// MoELayer_86371792323170
// MI455X (gfx1250) — matrix operand order repaired and checked by reading, not compiled, not run on hardware
//
#include <hip/hip_runtime.h>

#define D_MODEL 1024
#define D_FF    4096
#define NEXP    8

typedef __attribute__((ext_vector_type(16))) __bf16 v16bf_t;
typedef __attribute__((ext_vector_type(8)))  float  v8f_t;

union FragU {
  v16bf_t v;
  float4  q[2];
};

__device__ __forceinline__ unsigned short f2bf(float f) {
  unsigned int u = __float_as_uint(f);
  u += 0x7FFFu + ((u >> 16) & 1u);          // round-to-nearest-even
  return (unsigned short)(u >> 16);
}

// GELU (tanh approx) using the gfx1250 hardware transcendental v_tanh_f32.
// Two v_nops cover the TRANS-op RAW hazard window (ISA 01 §7.4.3).
__device__ __forceinline__ float gelu_f(float x) {
  const float c0 = 0.7978845608028654f;     // sqrt(2/pi)
  const float c1 = 0.044715f;
  float t = c0 * (x + c1 * x * x * x);
  float th;
  asm("v_tanh_f32 %0, %1\n\t"
      "v_nop\n\t"
      "v_nop"
      : "=v"(th) : "v"(t));
  return 0.5f * x * (1.0f + th);
}

// LDS byte offset of a pointer into __shared__ (flat LDS aperture: addr[31:0])
__device__ __forceinline__ unsigned lds_off_of(const void* p) {
  return (unsigned)(size_t)p;
}

// 64 contiguous bytes global -> LDS via async copy engine (4 x b128).
// offset: immediate is added to BOTH the LDS and the global address (ISA 08 §4.4).
__device__ __forceinline__ void async_copy64(unsigned lds_off,
                                             const unsigned short* base,
                                             unsigned byte_off) {
  asm volatile(
      "global_load_async_to_lds_b128 %0, %1, %2 offset:0\n\t"
      "global_load_async_to_lds_b128 %0, %1, %2 offset:16\n\t"
      "global_load_async_to_lds_b128 %0, %1, %2 offset:32\n\t"
      "global_load_async_to_lds_b128 %0, %1, %2 offset:48"
      :: "v"(lds_off), "v"(byte_off), "s"(base)
      : "memory");
}

__device__ __forceinline__ void wait_async8() {
  asm volatile("s_wait_asynccnt 0x8" ::: "memory");
}
__device__ __forceinline__ void wait_async0() {
  asm volatile("s_wait_asynccnt 0x0" ::: "memory");
}

// ---------------------------------------------------------------------------
// f32 -> bf16 conversion (8 elements / thread)
// ---------------------------------------------------------------------------
__global__ void cvt_bf16_kernel(const float* __restrict__ in,
                                uint4* __restrict__ out, int n8) {
  const int i = blockIdx.x * blockDim.x + threadIdx.x;
  if (i >= n8) return;
  const float4* s = (const float4*)(in + (size_t)i * 8);
  float4 a = s[0], b = s[1];
  uint4 o;
  o.x = (unsigned)f2bf(a.x) | ((unsigned)f2bf(a.y) << 16);
  o.y = (unsigned)f2bf(a.z) | ((unsigned)f2bf(a.w) << 16);
  o.z = (unsigned)f2bf(b.x) | ((unsigned)f2bf(b.y) << 16);
  o.w = (unsigned)f2bf(b.z) | ((unsigned)f2bf(b.w) << 16);
  out[i] = o;
}

// ---------------------------------------------------------------------------
// Weight transpose + convert: W f32 [K][N] (per expert) -> Wt bf16 [N][K]
// ---------------------------------------------------------------------------
__global__ void transcvt_kernel(const float* __restrict__ in,
                                unsigned short* __restrict__ out,
                                int K, int N) {
  __shared__ unsigned short tile[64 * 66];
  const size_t mat = (size_t)K * N;
  const float* src = in + (size_t)blockIdx.z * mat;
  unsigned short* dst = out + (size_t)blockIdx.z * mat;
  const int k0 = blockIdx.y * 64;
  const int n0 = blockIdx.x * 64;
  const int c = threadIdx.x & 63;
  const int r = threadIdx.x >> 6;  // 0..3
#pragma unroll
  for (int j = 0; j < 16; ++j) {
    const int kk = r + (j << 2);
    const float v = src[(size_t)(k0 + kk) * N + n0 + c];  // c = n (coalesced)
    tile[c * 66 + kk] = f2bf(v);                          // store [n][k]
  }
  __syncthreads();
#pragma unroll
  for (int j = 0; j < 16; ++j) {
    const int nn = r + (j << 2);
    dst[(size_t)(n0 + nn) * K + k0 + c] = tile[nn * 66 + c];  // c = k (coalesced)
  }
}

// ---------------------------------------------------------------------------
// Router: one wave32 per token. logits -> softmax -> top-2
// ---------------------------------------------------------------------------
__global__ void router_kernel(const float* __restrict__ h,
                              const float* __restrict__ Wr,
                              const float* __restrict__ br,
                              int* __restrict__ top_i,
                              float* __restrict__ top_v,
                              float* __restrict__ gsum,
                              int T) {
  const int lane = threadIdx.x & 31;
  const int w    = threadIdx.x >> 5;
  const int t    = blockIdx.x * 8 + w;
  if (t >= T) return;

  float acc[NEXP];
#pragma unroll
  for (int e = 0; e < NEXP; ++e) acc[e] = 0.f;

  const float* hr = h + (size_t)t * D_MODEL;
  for (int d = lane; d < D_MODEL; d += 32) {
    const float hv = hr[d];
    const float4* wr = (const float4*)(Wr + (size_t)d * NEXP);
    float4 w0 = wr[0], w1 = wr[1];
    acc[0] += hv * w0.x; acc[1] += hv * w0.y;
    acc[2] += hv * w0.z; acc[3] += hv * w0.w;
    acc[4] += hv * w1.x; acc[5] += hv * w1.y;
    acc[6] += hv * w1.z; acc[7] += hv * w1.w;
  }
#pragma unroll
  for (int e = 0; e < NEXP; ++e) {
#pragma unroll
    for (int off = 16; off > 0; off >>= 1)
      acc[e] += __shfl_xor(acc[e], off, 32);
  }
  if (lane == 0) {
    float mx = -3.0e38f;
#pragma unroll
    for (int e = 0; e < NEXP; ++e) { acc[e] += br[e]; mx = fmaxf(mx, acc[e]); }
    float s = 0.f;
#pragma unroll
    for (int e = 0; e < NEXP; ++e) { acc[e] = __expf(acc[e] - mx); s += acc[e]; }
    const float inv = 1.0f / s;
    int i0 = 0;
#pragma unroll
    for (int e = 1; e < NEXP; ++e) if (acc[e] > acc[i0]) i0 = e;
    int i1 = (i0 == 0) ? 1 : 0;
#pragma unroll
    for (int e = 0; e < NEXP; ++e) {
      if (e == i0) continue;
      if (acc[e] > acc[i1]) i1 = e;
    }
    top_i[2 * t]     = i0; top_v[2 * t]     = acc[i0] * inv;
    top_i[2 * t + 1] = i1; top_v[2 * t + 1] = acc[i1] * inv;
    float tot = 0.f;
#pragma unroll
    for (int e = 0; e < NEXP; ++e) tot += acc[e] * inv;
    atomicAdd(gsum, tot);
  }
}

// ---------------------------------------------------------------------------
// Compact top-2 assignments into contiguous per-expert slot ranges
// ---------------------------------------------------------------------------
__global__ void compact_kernel(const int* __restrict__ top_i,
                               int* __restrict__ list,
                               int* __restrict__ slot_of,
                               int* __restrict__ counts,
                               int* __restrict__ offs,
                               int T) {
  __shared__ int cnt[NEXP], base[NEXP], cur[NEXP];
  if (threadIdx.x < NEXP) { cnt[threadIdx.x] = 0; cur[threadIdx.x] = 0; }
  __syncthreads();
  for (int i = threadIdx.x; i < 2 * T; i += blockDim.x)
    atomicAdd(&cnt[top_i[i]], 1);
  __syncthreads();
  if (threadIdx.x == 0) {
    int o = 0;
    for (int e = 0; e < NEXP; ++e) { base[e] = o; o += cnt[e]; }
  }
  __syncthreads();
  for (int i = threadIdx.x; i < 2 * T; i += blockDim.x) {
    const int e = top_i[i];
    const int pos = base[e] + atomicAdd(&cur[e], 1);
    list[pos]  = i >> 1;
    slot_of[i] = pos;
  }
  __syncthreads();
  if (threadIdx.x < NEXP) {
    counts[threadIdx.x] = cnt[threadIdx.x];
    offs[threadIdx.x]   = base[threadIdx.x];
  }
}

// ---------------------------------------------------------------------------
// Tiled WMMA GEMM with async double-buffered LDS staging.
//   C[M,N] = act(A_bf16[M,K] @ Bt_bf16[N][K]^T + bias)
//   block tile 128x128, K-step 64 (two WMMA substeps), 8 waves, each wave a
//   64x32 region = 4x2 grid of v_wmma_f32_16x16x32_bf16 tiles.
//   Steady-state loop is branch-free; last iteration is peeled.
// ---------------------------------------------------------------------------
template <bool GATHER, bool GELU, bool OUT_BF16>
__global__ __launch_bounds__(256)
void gemm_wmma_kernel(const unsigned short* __restrict__ A, int lda,
                      const unsigned short* __restrict__ Bt,  // [N][K] bf16
                      int N,
                      const float* __restrict__ bias,
                      void* __restrict__ Out, int ldo,
                      const int* __restrict__ gather_list,
                      const int* __restrict__ counts,
                      const int* __restrict__ offs,
                      int K, int Mfixed) {
  constexpr int TM = 128, TN = 128, TK = 64, LS = 72;  // LDS row stride (elems)
  __shared__ __align__(16) unsigned short Asm[2][TM * LS];
  __shared__ __align__(16) unsigned short Bsm[2][TN * LS];

  int m0, mEnd;
  const unsigned short* Btp = Bt;
  const float* biasp = bias;
  if (counts != nullptr) {
    const int e   = blockIdx.z;
    const int cnt = counts[e];
    if ((int)blockIdx.y * TM >= cnt) return;
    const int off = offs[e];
    m0   = off + blockIdx.y * TM;
    mEnd = off + cnt;
    Btp   = Bt + (size_t)e * (size_t)K * (size_t)N;
    biasp = bias + (size_t)e * (size_t)N;
  } else {
    m0   = blockIdx.y * TM;
    mEnd = Mfixed;
  }
  const int n0 = blockIdx.x * TN;

  const int tid  = threadIdx.x;
  const int lane = tid & 31;
  const int wid  = tid >> 5;
  const int wm   = wid & 1;    // 64-row slab
  const int wn   = wid >> 1;   // 32-col slab

  v8f_t acc[4][2];
#pragma unroll
  for (int mi = 0; mi < 4; ++mi)
#pragma unroll
    for (int ni = 0; ni < 2; ++ni)
#pragma unroll
      for (int r = 0; r < 8; ++r) acc[mi][ni][r] = 0.0f;

  // staging: thread -> (row 0..127, 32-element half); rows past mEnd are
  // clamped to a valid row (their outputs are masked in the epilogue).
  const int crow  = tid >> 1;
  const int chalf = tid & 1;
  int asrc;
  {
    int m = m0 + crow;
    if (m >= mEnd) m = mEnd - 1;
    asrc = GATHER ? gather_list[m] : m;
  }
  const unsigned abyte = (unsigned)(((size_t)asrc * lda + chalf * 32) * 2);
  const unsigned bbyte = (unsigned)(((size_t)(n0 + crow) * K + chalf * 32) * 2);
  unsigned lds_a[2], lds_b[2];
#pragma unroll
  for (int b = 0; b < 2; ++b) {
    lds_a[b] = lds_off_of(&Asm[b][crow * LS + chalf * 32]);
    lds_b[b] = lds_off_of(&Bsm[b][crow * LS + chalf * 32]);
  }

  // one 64-deep compute step on buffer `buf`
  auto compute = [&](int buf) {
#pragma unroll
    for (int kk = 0; kk < 2; ++kk) {
      v16bf_t afrag[4];
#pragma unroll
      for (int mi = 0; mi < 4; ++mi) {
        // A frag (16x32): lane holds M=lane%16, K-runs {c..c+7} u {c+16..c+23}
        const int row = wm * 64 + mi * 16 + (lane & 15);
        const int c   = kk * 32 + ((lane & 16) ? 8 : 0);
        const unsigned short* p = &Asm[buf][row * LS + c];
        FragU f;
        f.q[0] = *(const float4*)(p);
        f.q[1] = *(const float4*)(p + 16);
        afrag[mi] = f.v;
      }
      v16bf_t bfrag[2];
#pragma unroll
      for (int ni = 0; ni < 2; ++ni) {
        // B frag (32x16): lane holds N=lane%16, K-runs {c..c+7} u {c+16..c+23}
        const int row = wn * 32 + ni * 16 + (lane & 15);
        const int c   = kk * 32 + ((lane & 16) ? 8 : 0);
        const unsigned short* p = &Bsm[buf][row * LS + c];
        FragU f;
        f.q[0] = *(const float4*)(p);
        f.q[1] = *(const float4*)(p + 16);
        bfrag[ni] = f.v;
      }
#pragma unroll
      for (int mi = 0; mi < 4; ++mi)
#pragma unroll
        for (int ni = 0; ni < 2; ++ni)
          acc[mi][ni] = __builtin_amdgcn_wmma_f32_16x16x32_bf16(
              false, afrag[mi], false, bfrag[ni], (short)0, acc[mi][ni], false, false);
    }
  };

  // prologue: stage tile 0 into buffer 0
  async_copy64(lds_a[0], A,   abyte);
  async_copy64(lds_b[0], Btp, bbyte);

  const int nIter = K / TK;  // >= 16 for all our shapes
  int it = 0;
  for (; it < nIter - 1; ++it) {
    const int buf = it & 1;
    const unsigned kb = (unsigned)((it + 1) * TK * 2);
    async_copy64(lds_a[buf ^ 1], A,   abyte + kb);
    async_copy64(lds_b[buf ^ 1], Btp, bbyte + kb);
    wait_async8();     // retire current tile's 8 chunks (in-order completion)
    __syncthreads();   // current buffer visible block-wide
    compute(buf);
    __syncthreads();   // done reading before this buffer is re-staged
  }
  // peeled last iteration
  wait_async0();
  __syncthreads();
  compute(it & 1);

  // ---- epilogue: bias (+GELU) and store ----
  const int laneN = lane & 15;
  const int laneM = (lane & 16) ? 8 : 0;
#pragma unroll
  for (int ni = 0; ni < 2; ++ni) {
    const int ng = n0 + wn * 32 + ni * 16 + laneN;
    const float bv = biasp[ng];
#pragma unroll
    for (int mi = 0; mi < 4; ++mi) {
#pragma unroll
      for (int r = 0; r < 8; ++r) {
        const int mg = m0 + wm * 64 + mi * 16 + r + laneM;
        if (mg < mEnd) {
          float v = acc[mi][ni][r] + bv;
          if (GELU) v = gelu_f(v);
          if (OUT_BF16)
            ((unsigned short*)Out)[(size_t)mg * ldo + ng] = f2bf(v);
          else
            ((float*)Out)[(size_t)mg * ldo + ng] = v;
        }
      }
    }
  }
}

// ---------------------------------------------------------------------------
// Combine: out[t] += w0 * eout[slot0[t]] + w1 * eout[slot1[t]]
// ---------------------------------------------------------------------------
__global__ void combine_kernel(float* __restrict__ out,
                               const float* __restrict__ eout,
                               const int* __restrict__ slot_of,
                               const float* __restrict__ top_v,
                               int T) {
  const int i = blockIdx.x * blockDim.x + threadIdx.x;  // float4 index
  const int t = i >> 8;                                 // 256 float4 per row
  if (t >= T) return;
  const int d4 = i & 255;
  const float w0 = top_v[2 * t], w1 = top_v[2 * t + 1];
  const int s0 = slot_of[2 * t], s1 = slot_of[2 * t + 1];
  const float4 a = ((const float4*)(eout + (size_t)s0 * D_MODEL))[d4];
  const float4 b = ((const float4*)(eout + (size_t)s1 * D_MODEL))[d4];
  float4* op = ((float4*)(out + (size_t)t * D_MODEL)) + d4;
  float4 v = *op;
  v.x += w0 * a.x + w1 * b.x;
  v.y += w0 * a.y + w1 * b.y;
  v.z += w0 * a.z + w1 * b.z;
  v.w += w0 * a.w + w1 * b.w;
  *op = v;
}

__global__ void init_kernel(float* gsum) {
  if (threadIdx.x == 0) gsum[0] = 0.f;
}

__global__ void finalize_kernel(float* __restrict__ out,
                                const float* __restrict__ gsum,
                                unsigned long long pos, float scale) {
  if (threadIdx.x == 0 && blockIdx.x == 0) out[pos] = gsum[0] * scale;
}

// ---------------------------------------------------------------------------
extern "C" void kernel_launch(void* const* d_in, const int* in_sizes, int n_in,
                              void* d_out, int out_size, void* d_ws, size_t ws_size,
                              hipStream_t stream) {
  (void)n_in; (void)out_size; (void)ws_size;
  const float* h   = (const float*)d_in[0];
  const float* Wr  = (const float*)d_in[1];
  const float* br  = (const float*)d_in[2];
  const float* We1 = (const float*)d_in[3];
  const float* be1 = (const float*)d_in[4];
  const float* We2 = (const float*)d_in[5];
  const float* be2 = (const float*)d_in[6];
  const float* Ws1 = (const float*)d_in[7];
  const float* bs1 = (const float*)d_in[8];
  const float* Ws2 = (const float*)d_in[9];
  const float* bs2 = (const float*)d_in[10];
  float* out = (float*)d_out;

  const int T = in_sizes[0] / D_MODEL;  // tokens (4096)
  const int S = 2 * T;                  // top-2 slots

  // workspace carve (all 256B aligned)
  char* w = (char*)d_ws;
  auto carve = [&](size_t bytes) -> char* {
    char* p = w;
    w += (bytes + 255) & ~(size_t)255;
    return p;
  };
  unsigned short* h_bf  = (unsigned short*)carve((size_t)T * D_MODEL * 2);
  unsigned short* hid   = (unsigned short*)carve((size_t)S * D_FF * 2);
  float* eout   = (float*)carve((size_t)S * D_MODEL * 4);
  unsigned short* wt_e1 = (unsigned short*)carve((size_t)NEXP * D_MODEL * D_FF * 2);
  unsigned short* wt_e2 = (unsigned short*)carve((size_t)NEXP * D_FF * D_MODEL * 2);
  unsigned short* wt_s1 = (unsigned short*)carve((size_t)D_MODEL * D_FF * 2);
  unsigned short* wt_s2 = (unsigned short*)carve((size_t)D_FF * D_MODEL * 2);
  int*   top_i   = (int*)carve((size_t)S * 4);
  float* top_v   = (float*)carve((size_t)S * 4);
  int*   slot_of = (int*)carve((size_t)S * 4);
  int*   list    = (int*)carve((size_t)S * 4);
  int*   counts  = (int*)carve(256);
  int*   offs    = (int*)carve(256);
  float* gsum    = (float*)carve(256);

  init_kernel<<<1, 32, 0, stream>>>(gsum);

  const int n8 = T * D_MODEL / 8;
  cvt_bf16_kernel<<<(n8 + 255) / 256, 256, 0, stream>>>(h, (uint4*)h_bf, n8);

  // pre-transpose + bf16-convert all weights: W[K][N] -> Wt[N][K]
  transcvt_kernel<<<dim3(D_FF / 64, D_MODEL / 64, NEXP), 256, 0, stream>>>(We1, wt_e1, D_MODEL, D_FF);
  transcvt_kernel<<<dim3(D_MODEL / 64, D_FF / 64, NEXP), 256, 0, stream>>>(We2, wt_e2, D_FF, D_MODEL);
  transcvt_kernel<<<dim3(D_FF / 64, D_MODEL / 64, 1),    256, 0, stream>>>(Ws1, wt_s1, D_MODEL, D_FF);
  transcvt_kernel<<<dim3(D_MODEL / 64, D_FF / 64, 1),    256, 0, stream>>>(Ws2, wt_s2, D_FF, D_MODEL);

  router_kernel<<<(T + 7) / 8, 256, 0, stream>>>(h, Wr, br, top_i, top_v, gsum, T);
  compact_kernel<<<1, 256, 0, stream>>>(top_i, list, slot_of, counts, offs, T);

  const dim3 blk(256);
  {  // expert up-proj + GELU: hid[slot] = gelu(h[tok] @ We1[e] + be1[e])
    dim3 g(D_FF / 128, T / 128, NEXP);
    gemm_wmma_kernel<true, true, true><<<g, blk, 0, stream>>>(
        h_bf, D_MODEL, wt_e1, D_FF, be1, hid, D_FF, list, counts, offs, D_MODEL, 0);
  }
  {  // expert down-proj: eout[slot] = hid[slot] @ We2[e] + be2[e]
    dim3 g(D_MODEL / 128, T / 128, NEXP);
    gemm_wmma_kernel<false, false, false><<<g, blk, 0, stream>>>(
        hid, D_FF, wt_e2, D_MODEL, be2, eout, D_MODEL, nullptr, counts, offs, D_FF, 0);
  }
  {  // shared up-proj + GELU (reuses hid buffer)
    dim3 g(D_FF / 128, T / 128, 1);
    gemm_wmma_kernel<false, true, true><<<g, blk, 0, stream>>>(
        h_bf, D_MODEL, wt_s1, D_FF, bs1, hid, D_FF, nullptr, nullptr, nullptr, D_MODEL, T);
  }
  {  // shared down-proj -> writes d_out directly
    dim3 g(D_MODEL / 128, T / 128, 1);
    gemm_wmma_kernel<false, false, false><<<g, blk, 0, stream>>>(
        hid, D_FF, wt_s2, D_MODEL, bs2, (void*)out, D_MODEL, nullptr, nullptr, nullptr, D_FF, T);
  }

  combine_kernel<<<(T * D_MODEL / 4 + 255) / 256, 256, 0, stream>>>(
      out, eout, slot_of, top_v, T);

  finalize_kernel<<<1, 32, 0, stream>>>(out, gsum, (unsigned long long)T * D_MODEL,
                                        1.0f / ((float)T * NEXP));
}
